// ALiBiAttention_6906307412356
// MI455X (gfx1250) — matrix operand order repaired and checked by reading, not compiled, not run on hardware
//
#include <hip/hip_runtime.h>
#include <hip/hip_bf16.h>

typedef __attribute__((ext_vector_type(16))) _Float16 v16h;
typedef __attribute__((ext_vector_type(8)))  _Float16 v8h;
typedef __attribute__((ext_vector_type(8)))  float    v8f;

#define V16CAT(lo, hi) __builtin_shufflevector((lo), (hi), 0,1,2,3,4,5,6,7,8,9,10,11,12,13,14,15)

static __device__ __forceinline__ v8f wmma_f16(v16h a, v16h b, v8f c) {
    return __builtin_amdgcn_wmma_f32_16x16x32_f16(false, a, false, b, (short)0, c, false, false);
}

// ---------------------------------------------------------------------------
// XOR-shuffle within 16-lane rows. Prefer v_permlane16_b32 (pure VALU, no LDS
// round-trip / s_wait_dscnt) with compile-time permutation constants.
// ---------------------------------------------------------------------------
template<int D, unsigned SEL0, unsigned SEL1>
static __device__ __forceinline__ float xor16(float x) {
#if __has_builtin(__builtin_amdgcn_permlane16)
    unsigned u = __float_as_uint(x);
    u = __builtin_amdgcn_permlane16(u, u, SEL0, SEL1, false, false);
    return __uint_as_float(u);
#else
    return __shfl_xor(x, D, 16);
#endif
}

// reduce x across the 16-lane row with op (max or add)
#define RED16MAX(x) do { \
    x = fmaxf(x, xor16<1, 0x67452301u, 0xEFCDAB89u>(x)); \
    x = fmaxf(x, xor16<2, 0x54761032u, 0xDCFE98BAu>(x)); \
    x = fmaxf(x, xor16<4, 0x32107654u, 0xBA98FEDCu>(x)); \
    x = fmaxf(x, xor16<8, 0xFEDCBA98u, 0x76543210u>(x)); \
} while (0)
#define RED16SUM(x) do { \
    x += xor16<1, 0x67452301u, 0xEFCDAB89u>(x); \
    x += xor16<2, 0x54761032u, 0xDCFE98BAu>(x); \
    x += xor16<4, 0x32107654u, 0xBA98FEDCu>(x); \
    x += xor16<8, 0xFEDCBA98u, 0x76543210u>(x); \
} while (0)

// ---------------------------------------------------------------------------
// CDNA5 async global -> LDS copy (16 bytes), tracked by ASYNCcnt.
// GV mode: vdst = per-lane LDS byte address, vaddr = 64-bit global address.
// LDS byte address = low 32 bits of the flat shared pointer (ISA 10.2).
// ---------------------------------------------------------------------------
static __device__ __forceinline__ void g2l_async_b128(const _Float16* g, _Float16* l) {
    unsigned           lds = (unsigned)(size_t)l;
    unsigned long long ga  = (unsigned long long)(size_t)g;
    asm volatile("global_load_async_to_lds_b128 %0, %1, off"
                 :: "v"(lds), "v"(ga) : "memory");
}
static __device__ __forceinline__ void wait_async0() {
    asm volatile("s_wait_asynccnt 0x0" ::: "memory");
}

// ---------------------------------------------------------------------------
// f32 -> f16 conversion
// ---------------------------------------------------------------------------
__global__ void cvt_f32_f16(const float* __restrict__ src, _Float16* __restrict__ dst, int n) {
    int i = blockIdx.x * blockDim.x + threadIdx.x;
    if (i < n) dst[i] = (_Float16)src[i];
}

// ---------------------------------------------------------------------------
// C[M,N] = A[M,K] @ B[N,K]^T + bias[N]
// block = 256 threads = 8 waves; block tile 128x64; K-step 32.
// A/B tiles staged to LDS with async-to-LDS DMA; wave (wid&3, wid>>2)
// computes a 32x32 sub-tile (2x2 WMMA tiles).
// ---------------------------------------------------------------------------
template<bool OUT_F16>
__global__ void gemm_wmma_kernel(const _Float16* __restrict__ A,
                                 const _Float16* __restrict__ Bm,
                                 const float*    __restrict__ bias,
                                 void* __restrict__ Cout,
                                 int M, int N, int K) {
    const int tid  = threadIdx.x;
    const int lane = tid & 31;
    const int wid  = tid >> 5;
    const int hl   = lane >> 4;    // lane half (0/1)
    const int lm   = lane & 15;

    const int m0 = blockIdx.y * 128;
    const int n0 = blockIdx.x * 64;
    const int wm = (wid & 3) * 32;
    const int wn = (wid >> 2) * 32;

    __shared__ __align__(16) _Float16 As[128 * 32];
    __shared__ __align__(16) _Float16 Bs[64 * 32];

    v8f acc[2][2] = {};

    for (int k0 = 0; k0 < K; k0 += 32) {
        __syncthreads();
        {   // A tile: 128x32 f16, 16 elems (2 x async b128) per thread
            int idx = tid * 16;
            int row = idx >> 5, kk = idx & 31;
            const _Float16* src = A + (size_t)(m0 + row) * K + k0 + kk;
            _Float16* dst = As + row * 32 + kk;
            g2l_async_b128(src,     dst);
            g2l_async_b128(src + 8, dst + 8);
        }
        {   // B tile: 64x32 f16, 8 elems (1 x async b128) per thread
            int idx = tid * 8;
            int row = idx >> 5, kk = idx & 31;
            g2l_async_b128(Bm + (size_t)(n0 + row) * K + k0 + kk,
                           Bs + row * 32 + kk);
        }
        wait_async0();
        __syncthreads();

        v16h afrag[2], bfrag[2];
#pragma unroll
        for (int i = 0; i < 2; ++i) {
            // A fragment: lane hl=0 -> K 0-7 & 16-23 ; hl=1 -> K 8-15 & 24-31
            const _Float16* ap = As + (wm + i * 16 + lm) * 32 + hl * 8;
            v8h lo = *(const v8h*)(ap);
            v8h hi = *(const v8h*)(ap + 16);
            afrag[i] = V16CAT(lo, hi);
        }
#pragma unroll
        for (int j = 0; j < 2; ++j) {
            // B fragment: lane hl=0 -> K 0-7 & 16-23 ; hl=1 -> K 8-15 & 24-31, col = lm
            const _Float16* bp = Bs + (wn + j * 16 + lm) * 32 + hl * 8;
            v8h lo = *(const v8h*)(bp);
            v8h hi = *(const v8h*)(bp + 16);
            bfrag[j] = V16CAT(lo, hi);
        }
#pragma unroll
        for (int i = 0; i < 2; ++i)
#pragma unroll
            for (int j = 0; j < 2; ++j)
                acc[i][j] = wmma_f16(afrag[i], bfrag[j], acc[i][j]);
    }

    // Epilogue: C layout -> row = base + hl*8 + r, col = base + lm
#pragma unroll
    for (int i = 0; i < 2; ++i) {
#pragma unroll
        for (int j = 0; j < 2; ++j) {
            int col = n0 + wn + j * 16 + lm;
            float bv = bias[col];
#pragma unroll
            for (int r = 0; r < 8; ++r) {
                int row = m0 + wm + i * 16 + hl * 8 + r;
                float v = acc[i][j][r] + bv;
                if (OUT_F16)
                    ((_Float16*)Cout)[(size_t)row * N + col] = (_Float16)v;
                else
                    ((float*)Cout)[(size_t)row * N + col] = v;
            }
        }
    }
}

// ---------------------------------------------------------------------------
// Flash attention with causal + ALiBi bias.
// qkv: [B*S, 3*D] f16 (q | k | v per row).  attn: [B*S, D] f16.
// One wave owns 16 query rows; 8 waves/block; grid = B*H*(S/128).
// ---------------------------------------------------------------------------
__global__ void attn_alibi_kernel(const _Float16* __restrict__ qkv,
                                  _Float16* __restrict__ attn) {
    const int S  = 2048;
    const int D  = 1024;
    const int D3 = 3072;
    const int HD = 64;
    const int H  = 16;

    const int tid  = threadIdx.x;
    const int lane = tid & 31;
    const int wid  = tid >> 5;
    const int hl   = lane >> 4;
    const int lm   = lane & 15;

    const int qt = blockIdx.x & 15;          // 16 q-tiles of 128 rows
    const int bh = blockIdx.x >> 4;
    const int h  = bh % H;
    const int b  = bh / H;
    const int qbase = qt * 128 + wid * 16;

    __shared__ __align__(16) _Float16 Plds[8 * 16 * 32];
    _Float16* pl = Plds + wid * (16 * 32);

    const _Float16* qBase = qkv + ((size_t)b * S) * D3 + h * HD;
    const _Float16* kBase = qBase + D;       // +1024
    const _Float16* vBase = qBase + 2 * D;   // +2048

    // Q A-fragments for hd chunks 0-31 and 32-63 (loaded once)
    v16h qf[2];
    {
        const _Float16* qp = qBase + (size_t)(qbase + lm) * D3;
#pragma unroll
        for (int c = 0; c < 2; ++c) {
            const _Float16* p = qp + c * 32 + hl * 8;
            v8h lo = *(const v8h*)(p);
            v8h hi = *(const v8h*)(p + 16);
            qf[c] = V16CAT(lo, hi);
        }
    }

    v8f accO[4] = {};
    float mrow[8], lrow[8];
#pragma unroll
    for (int r = 0; r < 8; ++r) { mrow[r] = -1e30f; lrow[r] = 0.0f; }

    const int qend = qbase + 15;
    for (int kt = 0; kt <= qend; kt += 32) {
        if (kt + 32 <= qend)  // pull next K tile toward caches
            __builtin_prefetch(kBase + (size_t)(kt + 32) * D3, 0, 1);

        // ---- scores: Q(16x64) @ K^T -> two 16x16 tiles (key halves) ----
        v8f sc[2] = {};
#pragma unroll
        for (int t = 0; t < 2; ++t) {
            int kcol = kt + t * 16 + lm;     // this lane's key column
#pragma unroll
            for (int c = 0; c < 2; ++c) {
                const _Float16* kp = kBase + (size_t)kcol * D3 + c * 32 + hl * 8;
                v8h lo = *(const v8h*)(kp);
                v8h hi = *(const v8h*)(kp + 16);
                v16h bf = V16CAT(lo, hi);
                sc[t] = wmma_f16(qf[c], bf, sc[t]);
            }
        }

        // ---- scale + ALiBi causal bias + online softmax ----
        float p0[8], p1[8], alpha[8];
#pragma unroll
        for (int r = 0; r < 8; ++r) {
            int q  = qbase + hl * 8 + r;
            int k0 = kt + lm;
            int k1 = kt + 16 + lm;
            float s0 = sc[0][r] * 0.125f + ((k0 <= q) ? (float)(k0 - q) : -1e30f);
            float s1 = sc[1][r] * 0.125f + ((k1 <= q) ? (float)(k1 - q) : -1e30f);
            float mx = fmaxf(s0, s1);
            RED16MAX(mx);
            float mnew = fmaxf(mrow[r], mx);
            float a  = __expf(mrow[r] - mnew);
            float e0 = __expf(s0 - mnew);
            float e1 = __expf(s1 - mnew);
            float rs = e0 + e1;
            RED16SUM(rs);
            lrow[r] = lrow[r] * a + rs;
            mrow[r] = mnew;
            alpha[r] = a;
            p0[r] = e0;
            p1[r] = e1;
        }

        // rescale running output accumulators
#pragma unroll
        for (int j = 0; j < 4; ++j)
#pragma unroll
            for (int r = 0; r < 8; ++r)
                accO[j][r] *= alpha[r];

        // ---- reshape P (C-layout) -> A-fragment via per-wave LDS slice ----
#pragma unroll
        for (int r = 0; r < 8; ++r) {
            pl[(hl * 8 + r) * 32 + lm]      = (_Float16)p0[r];
            pl[(hl * 8 + r) * 32 + lm + 16] = (_Float16)p1[r];
        }
        asm volatile("s_wait_dscnt 0x0" ::: "memory");
        v16h pf;
        {
            const _Float16* pp = pl + lm * 32 + hl * 16;
            v8h lo = *(const v8h*)(pp);
            v8h hi = *(const v8h*)(pp + 8);
            pf = V16CAT(lo, hi);
        }

        // ---- P(16x32) @ V(32x64): 4 hd chunks of 16 cols ----
#pragma unroll
        for (int j = 0; j < 4; ++j) {
            const _Float16* vp = vBase + (size_t)(kt + hl * 16) * D3 + j * 16 + lm;
            v16h vf;
#pragma unroll
            for (int e = 0; e < 16; ++e)
                vf[e] = vp[(size_t)e * D3];
            accO[j] = wmma_f16(pf, vf, accO[j]);
        }
    }

    // ---- normalize and store f16 [B*S, D] (head-concatenated) ----
#pragma unroll
    for (int r = 0; r < 8; ++r) lrow[r] = 1.0f / lrow[r];
    _Float16* outBase = attn + ((size_t)b * S) * D + h * HD;
#pragma unroll
    for (int j = 0; j < 4; ++j) {
        int col = j * 16 + lm;
#pragma unroll
        for (int r = 0; r < 8; ++r) {
            int q = qbase + hl * 8 + r;
            outBase[(size_t)q * D + col] = (_Float16)(accO[j][r] * lrow[r]);
        }
    }
}

// ---------------------------------------------------------------------------
// Launcher
// ---------------------------------------------------------------------------
extern "C" void kernel_launch(void* const* d_in, const int* in_sizes, int n_in,
                              void* d_out, int out_size, void* d_ws, size_t ws_size,
                              hipStream_t stream) {
    const float* x     = (const float*)d_in[0];
    const float* w_qkv = (const float*)d_in[1];
    const float* b_qkv = (const float*)d_in[2];
    const float* w_out = (const float*)d_in[3];
    const float* b_out = (const float*)d_in[4];

    const int B = 4, S = 2048, D = 1024, H = 16;
    const size_t M = (size_t)B * S;   // 8192

    _Float16* xF    = (_Float16*)d_ws;
    _Float16* wqkvF = xF    + M * D;
    _Float16* woutF = wqkvF + (size_t)3 * D * D;
    _Float16* qkvF  = woutF + (size_t)D * D;
    _Float16* attnF = qkvF  + M * 3 * D;

    {   // f32 -> f16 conversions
        size_t n;
        n = M * D;
        cvt_f32_f16<<<dim3((unsigned)((n + 255) / 256)), dim3(256), 0, stream>>>(x, xF, (int)n);
        n = (size_t)3 * D * D;
        cvt_f32_f16<<<dim3((unsigned)((n + 255) / 256)), dim3(256), 0, stream>>>(w_qkv, wqkvF, (int)n);
        n = (size_t)D * D;
        cvt_f32_f16<<<dim3((unsigned)((n + 255) / 256)), dim3(256), 0, stream>>>(w_out, woutF, (int)n);
    }

    // QKV projection: [M, D] @ [3D, D]^T -> [M, 3D] f16
    gemm_wmma_kernel<true><<<dim3(3 * D / 64, (unsigned)(M / 128)), dim3(256), 0, stream>>>(
        xF, wqkvF, b_qkv, (void*)qkvF, (int)M, 3 * D, D);

    // Attention (causal + ALiBi), heads re-concatenated into [M, D] f16
    attn_alibi_kernel<<<dim3((unsigned)(B * H * (S / 128))), dim3(256), 0, stream>>>(qkvF, attnF);

    // Output projection: [M, D] @ [D, D]^T + b_out -> f32 d_out
    gemm_wmma_kernel<false><<<dim3(D / 64, (unsigned)(M / 128)), dim3(256), 0, stream>>>(
        attnF, woutF, b_out, d_out, (int)M, D, D);
}
